// stCMGAE_model_43276090474809
// MI455X (gfx1250) — matrix operand order repaired and checked by reading, not compiled, not run on hardware
//
#include <hip/hip_runtime.h>

// Problem constants (match reference)
#define N_NODES 10000
#define E_EDGES 60000
#define D_FEAT  3000
#define DPAD    3008   // D padded to multiple of 32 (zero-filled)
#define H_DIM   512
#define L_DIM   128
#define P_DIM   256
#define NMASK   3000

// 1 = stage GEMM tiles with GLOBAL_LOAD_ASYNC_TO_LDS_B128 (ASYNCcnt path,
//     LDS double-buffered); 0 = register-pipelined global->reg->LDS path.
#define USE_ASYNC_LDS 1

typedef __bf16 bf16_t;
typedef __attribute__((ext_vector_type(16))) __bf16 v16bf;
typedef __attribute__((ext_vector_type(8)))  __bf16 v8bf;
typedef __attribute__((ext_vector_type(8)))  float   v8f;

__device__ inline bf16_t f2bf(float f) {
  union { float f; unsigned u; } v; v.f = f;
  unsigned r = v.u + 0x7FFFu + ((v.u >> 16) & 1u);   // round-to-nearest-even
  unsigned short h = (unsigned short)(r >> 16);
  return __builtin_bit_cast(bf16_t, h);
}
__device__ inline float sigmoidf(float x) { return 1.0f / (1.0f + expf(-x)); }

// ---------------------------------------------------------------------------
// WMMA bf16 GEMM: C[M,Nc] = A[M,Ka] @ Bt[Nc,Ka]^T ; A row-major (stride Ka),
// Bt row-major transposed weights (stride Ka), C fp32 (stride Nc).
// Ka must be a multiple of 32 (callers pad with zeros).
// Block: 256 threads = 8 waves (wave32). C tile 64x128; wave -> 32x32 subtile.
// ---------------------------------------------------------------------------
__device__ inline v16bf make_frag(const bf16_t* p0, const bf16_t* p1) {
  v8bf lo = *(const v8bf*)p0;
  v8bf hi = *(const v8bf*)p1;
  v16bf r;
#pragma unroll
  for (int i = 0; i < 8; ++i) { r[i] = lo[i]; r[i + 8] = hi[i]; }
  return r;
}

#define WMMA_BF16(a, b, c) \
  __builtin_amdgcn_wmma_f32_16x16x32_bf16(false, (a), false, (b), (short)0, (c), false, false)

#if USE_ASYNC_LDS
// Per-lane async 16B copy: global -> LDS, tracked by ASYNCcnt.
__device__ inline void async_b128(unsigned ldsOff, const bf16_t* g) {
  asm volatile("global_load_async_to_lds_b128 %0, %1, off"
               :: "v"(ldsOff), "v"(g) : "memory");
}
__device__ inline void wait_async0() { asm volatile("s_wait_asynccnt 0x0" ::: "memory"); }
__device__ inline void wait_async3() { asm volatile("s_wait_asynccnt 0x3" ::: "memory"); }
#endif

__global__ __launch_bounds__(256) void gemm_bf16_k(const bf16_t* __restrict__ A,
                                                   const bf16_t* __restrict__ Bt,
                                                   float* __restrict__ C,
                                                   int M, int Nc, int Ka) {
#if USE_ASYNC_LDS
  __shared__ __align__(16) bf16_t Alds[2][64 * 32];    // [m][k], double-buffered
  __shared__ __align__(16) bf16_t Blds[2][128 * 32];   // [n][k], double-buffered
#else
  __shared__ __align__(16) bf16_t Alds[1][64 * 32];
  __shared__ __align__(16) bf16_t Blds[1][128 * 32];
#endif
  const int tid  = threadIdx.x;
  const int lane = tid & 31;
  const int wave = tid >> 5;
  const int wm   = wave >> 2;   // 0..1
  const int wn   = wave & 3;    // 0..3
  const int tileM = blockIdx.y * 64;
  const int tileN = blockIdx.x * 128;

  // Staging coordinates: A = 64x32 elems = 256 x v8bf (1/thread);
  // B = 128x32 elems = 512 x v8bf (2/thread). Clamp rows/cols (safe: clamped
  // lanes only feed C entries that the guarded store skips).
  const int arow = tid >> 2;
  const int akk  = (tid & 3) * 8;
  int gmA = tileM + arow; if (gmA > M - 1) gmA = M - 1;
  const size_t aBase = (size_t)gmA * Ka;

  const int bc0 = tid >> 2;            // cols 0..63
  const int bk0 = (tid & 3) * 8;
  const int bc1 = bc0 + 64;            // cols 64..127
  int gn0 = tileN + bc0; if (gn0 > Nc - 1) gn0 = Nc - 1;
  int gn1 = tileN + bc1; if (gn1 > Nc - 1) gn1 = Nc - 1;
  const size_t b0Base = (size_t)gn0 * Ka;
  const size_t b1Base = (size_t)gn1 * Ka;

  v8f c00 = {}, c01 = {}, c10 = {}, c11 = {};

  // Fragment coordinates (ISA 7.12.2 layouts).
  const int frow = lane & 15;
  const int kb   = (lane >> 4) * 8;     // A: K sub-block select
  const int koff = (lane >> 4) * 8;     // B: K sub-block select
  const int aFrag0 = (wm * 32 + frow) * 32 + kb;
  const int bFrag0 = (wn * 32 + frow) * 32 + koff;

#if USE_ASYNC_LDS
  const unsigned aDst = (unsigned)(size_t)&Alds[0][arow * 32 + akk];
  const unsigned b0Dst = (unsigned)(size_t)&Blds[0][bc0 * 32 + bk0];
  const unsigned b1Dst = (unsigned)(size_t)&Blds[0][bc1 * 32 + bk0];
  const unsigned aStep = (unsigned)(sizeof(bf16_t) * 64 * 32);
  const unsigned bStep = (unsigned)(sizeof(bf16_t) * 128 * 32);

  // Prologue: DMA tile 0 into buffer 0.
  async_b128(aDst,  &A[aBase + akk]);
  async_b128(b0Dst, &Bt[b0Base + bk0]);
  async_b128(b1Dst, &Bt[b1Base + bk0]);

  for (int k0 = 0; k0 < Ka; k0 += 32) {
    const int cur  = (k0 >> 5) & 1;
    const int kn   = k0 + 32;
    const bool more = kn < Ka;
    if (more) {
      const unsigned nb = (unsigned)(cur ^ 1);
      async_b128(aDst  + nb * aStep, &A[aBase + kn + akk]);
      async_b128(b0Dst + nb * bStep, &Bt[b0Base + kn + bk0]);
      async_b128(b1Dst + nb * bStep, &Bt[b1Base + kn + bk0]);
      __builtin_prefetch(&A[aBase + kn + 32 + akk], 0, 1);
      __builtin_prefetch(&Bt[b0Base + kn + 32 + bk0], 0, 1);
    }
    if (more) wait_async3(); else wait_async0();  // current tile's 3 ops done
    __syncthreads();

    const bf16_t* a0p = &Alds[cur][aFrag0];
    const bf16_t* b0p = &Blds[cur][bFrag0];
    v16bf a0 = make_frag(a0p, a0p + 16);
    v16bf a1 = make_frag(a0p + 16 * 32, a0p + 16 * 32 + 16);
    v16bf b0 = make_frag(b0p, b0p + 16);
    v16bf b1 = make_frag(b0p + 16 * 32, b0p + 16 * 32 + 16);

    c00 = WMMA_BF16(a0, b0, c00);
    c01 = WMMA_BF16(a0, b1, c01);
    c10 = WMMA_BF16(a1, b0, c10);
    c11 = WMMA_BF16(a1, b1, c11);
    __syncthreads();   // protect buf[cur] before it is re-targeted next iter
  }
#else
  v8bf aR  = *(const v8bf*)&A[aBase + akk];
  v8bf bR0 = *(const v8bf*)&Bt[b0Base + bk0];
  v8bf bR1 = *(const v8bf*)&Bt[b1Base + bk0];

  for (int k0 = 0; k0 < Ka; k0 += 32) {
    *(v8bf*)&Alds[0][arow * 32 + akk] = aR;
    *(v8bf*)&Blds[0][bc0 * 32 + bk0]  = bR0;
    *(v8bf*)&Blds[0][bc1 * 32 + bk0]  = bR1;
    __syncthreads();

    const int kn = k0 + 32;
    if (kn < Ka) {
      aR  = *(const v8bf*)&A[aBase + kn + akk];
      bR0 = *(const v8bf*)&Bt[b0Base + kn + bk0];
      bR1 = *(const v8bf*)&Bt[b1Base + kn + bk0];
      __builtin_prefetch(&A[aBase + kn + 32 + akk], 0, 1);
      __builtin_prefetch(&Bt[b0Base + kn + 32 + bk0], 0, 1);
    }

    const bf16_t* a0p = &Alds[0][aFrag0];
    const bf16_t* b0p = &Blds[0][bFrag0];
    v16bf a0 = make_frag(a0p, a0p + 16);
    v16bf a1 = make_frag(a0p + 16 * 32, a0p + 16 * 32 + 16);
    v16bf b0 = make_frag(b0p, b0p + 16);
    v16bf b1 = make_frag(b0p + 16 * 32, b0p + 16 * 32 + 16);
    __syncthreads();

    c00 = WMMA_BF16(a0, b0, c00);
    c01 = WMMA_BF16(a0, b1, c01);
    c10 = WMMA_BF16(a1, b0, c10);
    c11 = WMMA_BF16(a1, b1, c11);
  }
#endif

  // C/D layout: VGPR v -> M = v + 8*(lane>=16); N = lane&15.
  const int chi = (lane >> 4) * 8;
  const int cn  = lane & 15;
#pragma unroll
  for (int v = 0; v < 8; ++v) {
    int r0 = tileM + wm * 32 + chi + v;
    int r1 = r0 + 16;
    int n0 = tileN + wn * 32 + cn;
    int n1 = n0 + 16;
    if (r0 < M) {
      if (n0 < Nc) C[(size_t)r0 * Nc + n0] = c00[v];
      if (n1 < Nc) C[(size_t)r0 * Nc + n1] = c01[v];
    }
    if (r1 < M) {
      if (n0 < Nc) C[(size_t)r1 * Nc + n0] = c10[v];
      if (n1 < Nc) C[(size_t)r1 * Nc + n1] = c11[v];
    }
  }
}

// ---------------------------------------------------------------------------
// Conversion kernels
// ---------------------------------------------------------------------------
__global__ void convert_pad_k(const float* __restrict__ in, bf16_t* __restrict__ out,
                              long rows, int K, int Kp) {
  long total = rows * (long)Kp;
  for (long i = blockIdx.x * (long)blockDim.x + threadIdx.x; i < total;
       i += (long)gridDim.x * blockDim.x) {
    long r = i / Kp;
    int  k = (int)(i - r * Kp);
    out[i] = (k < K) ? f2bf(in[r * (long)K + k]) : f2bf(0.0f);
  }
}

// fp32 W (K x N) -> bf16 transposed Wt (N x Kp), zero pad K..Kp.
__global__ void convert_w_t_k(const float* __restrict__ in, bf16_t* __restrict__ out,
                              int K, int N, int Kp) {
  long total = (long)N * Kp;
  for (long i = blockIdx.x * (long)blockDim.x + threadIdx.x; i < total;
       i += (long)gridDim.x * blockDim.x) {
    int n = (int)(i / Kp);
    int k = (int)(i - (long)n * Kp);
    out[i] = (k < K) ? f2bf(in[(size_t)k * N + n]) : f2bf(0.0f);
  }
}

__global__ void convert_f2b_k(const float* __restrict__ in, bf16_t* __restrict__ out, long n) {
  for (long i = blockIdx.x * (long)blockDim.x + threadIdx.x; i < n;
       i += (long)gridDim.x * blockDim.x)
    out[i] = f2bf(in[i]);
}

__global__ void convert_posmask_k(const float* __restrict__ x, const int* __restrict__ perm,
                                  const float* __restrict__ pt, bf16_t* __restrict__ A) {
  int m = blockIdx.y;
  int f = blockIdx.x * 256 + threadIdx.x;
  if (f >= D_FEAT) return;
  int node = perm[m];
  A[(size_t)node * DPAD + f] = f2bf(x[(size_t)node * D_FEAT + f] + pt[f]);
}

__global__ void build_neg_k(const float* __restrict__ x, const int* __restrict__ perm,
                            const int* __restrict__ shuf, const float* __restrict__ nt,
                            bf16_t* __restrict__ A) {
  int row = blockIdx.y;
  int f = blockIdx.x * 256 + threadIdx.x;
  if (f >= DPAD) return;
  int node = perm[row];
  float v = 0.0f;
  if (f < D_FEAT) {
    if (row < NMASK) {
      v = nt[f];
    } else {
      int srcnode = perm[NMASK + shuf[row - NMASK]];
      v = x[(size_t)srcnode * D_FEAT + f];
    }
  }
  A[(size_t)node * DPAD + f] = f2bf(v);
}

// ---------------------------------------------------------------------------
// Graph kernels
// ---------------------------------------------------------------------------
__global__ void init_deg_k(float* deg) {
  int i = blockIdx.x * 256 + threadIdx.x;
  if (i < N_NODES) deg[i] = 1.0f;  // self loop
}
__global__ void deg_edges_k(const int* __restrict__ dst, float* deg) {
  int e = blockIdx.x * 256 + threadIdx.x;
  if (e < E_EDGES) atomicAdd(&deg[dst[e]], 1.0f);
}
__global__ void dinv_k(const float* deg, float* dinv) {
  int i = blockIdx.x * 256 + threadIdx.x;
  if (i < N_NODES) dinv[i] = rsqrtf(deg[i]);
}
__global__ void n2m_init_k(int* n2m) {
  int i = blockIdx.x * 256 + threadIdx.x;
  if (i < N_NODES) n2m[i] = -1;
}
__global__ void n2m_set_k(const int* perm, int* n2m) {
  int m = blockIdx.x * 256 + threadIdx.x;
  if (m < NMASK) n2m[perm[m]] = m;
}

// AGG[i] = XW[i] * dinv[i]^2  (covers self-loop edge)
__global__ void agg_init_k(const float* __restrict__ XW, const float* __restrict__ dinv,
                           float* __restrict__ AGG, int F) {
  int row = blockIdx.y;
  int f = blockIdx.x * 256 + threadIdx.x;
  if (f >= F) return;
  float w = dinv[row];
  AGG[(size_t)row * F + f] = XW[(size_t)row * F + f] * w * w;
}
__global__ void agg_edges_k(const float* __restrict__ XW, const int* __restrict__ src,
                            const int* __restrict__ dst, const float* __restrict__ dinv,
                            float* __restrict__ AGG, int F) {
  int e = blockIdx.y;
  int f = blockIdx.x * 256 + threadIdx.x;
  if (f >= F) return;
  int s = src[e], d = dst[e];
  float w = dinv[s] * dinv[d];
  atomicAdd(&AGG[(size_t)d * F + f], XW[(size_t)s * F + f] * w);
}

__global__ void bias_prelu_k(const float* __restrict__ in, const float* __restrict__ b,
                             const float* __restrict__ a, float* __restrict__ outf,
                             bf16_t* __restrict__ outb, int F, long total) {
  float alpha = a[0];
  for (long i = blockIdx.x * (long)blockDim.x + threadIdx.x; i < total;
       i += (long)gridDim.x * blockDim.x) {
    float v = in[i] + b[i % F];
    v = (v >= 0.0f) ? v : alpha * v;
    if (outf) outf[i] = v;
    outb[i] = f2bf(v);
  }
}

__global__ void add_bias_k(float* __restrict__ buf, const float* __restrict__ b, int F, long total) {
  for (long i = blockIdx.x * (long)blockDim.x + threadIdx.x; i < total;
       i += (long)gridDim.x * blockDim.x)
    buf[i] += b[i % F];
}

__global__ void gather_rows_k(const bf16_t* __restrict__ src, const int* __restrict__ perm,
                              bf16_t* __restrict__ dst, int F) {
  int m = blockIdx.y;
  int f = blockIdx.x * 256 + threadIdx.x;
  if (f >= F) return;
  dst[(size_t)m * F + f] = src[(size_t)perm[m] * F + f];
}

__global__ void zero_mask_rows_k(float* __restrict__ rec, const int* __restrict__ perm) {
  int m = blockIdx.x;
  rec[(size_t)perm[m] * L_DIM + threadIdx.x] = 0.0f;  // blockDim = 128
}

__global__ void rec_mask_init_k(const float* __restrict__ recW, const float* __restrict__ dinv,
                                const int* __restrict__ perm, float* __restrict__ rm) {
  int m = blockIdx.y;
  int f = blockIdx.x * 256 + threadIdx.x;
  if (f >= D_FEAT) return;
  int node = perm[m];
  float w = dinv[node];
  rm[(size_t)m * D_FEAT + f] = recW[(size_t)node * D_FEAT + f] * w * w;
}
__global__ void rec_mask_edges_k(const float* __restrict__ recW, const int* __restrict__ src,
                                 const int* __restrict__ dst, const int* __restrict__ n2m,
                                 const float* __restrict__ dinv, float* __restrict__ rm) {
  int e = blockIdx.y;
  int d = dst[e];
  int mi = n2m[d];
  if (mi < 0) return;
  int f = blockIdx.x * 256 + threadIdx.x;
  if (f >= D_FEAT) return;
  int s = src[e];
  float w = dinv[s] * dinv[d];
  atomicAdd(&rm[(size_t)mi * D_FEAT + f], recW[(size_t)s * D_FEAT + f] * w);
}

// ---------------------------------------------------------------------------
// Loss reductions
// ---------------------------------------------------------------------------
__global__ void colmean_sigmoid_k(const float* __restrict__ z, float* __restrict__ s,
                                  int rows, int cols) {
  int col = blockIdx.x;
  float acc = 0.0f;
  for (int r = threadIdx.x; r < rows; r += blockDim.x) acc += z[(size_t)r * cols + col];
  __shared__ float sm[256];
  sm[threadIdx.x] = acc;
  __syncthreads();
  for (int st = 128; st > 0; st >>= 1) {
    if ((int)threadIdx.x < st) sm[threadIdx.x] += sm[threadIdx.x + st];
    __syncthreads();
  }
  if (threadIdx.x == 0) s[col] = sigmoidf(sm[0] / (float)rows);
}

__global__ void matvec128_k(const float* __restrict__ W, const float* __restrict__ s,
                            float* __restrict__ out) {
  int r = threadIdx.x;  // 128 threads
  float acc = 0.0f;
  for (int c = 0; c < L_DIM; ++c) acc += W[r * L_DIM + c] * s[c];
  out[r] = acc;
}

__global__ void dgi_rows_k(const float* __restrict__ pz, const float* __restrict__ nz,
                           const float* __restrict__ wsv, float* __restrict__ acc) {
  int m = blockIdx.x;
  int c = threadIdx.x;  // 128
  __shared__ float smp[128], smn[128];
  float w = wsv[c];
  smp[c] = pz[(size_t)m * L_DIM + c] * w;
  smn[c] = nz[(size_t)m * L_DIM + c] * w;
  __syncthreads();
  for (int st = 64; st > 0; st >>= 1) {
    if (c < st) { smp[c] += smp[c + st]; smn[c] += smn[c + st]; }
    __syncthreads();
  }
  if (c == 0) {
    float pd = sigmoidf(smp[0]);
    float nd = sigmoidf(smn[0]);
    atomicAdd(&acc[1], -logf(pd + 1e-15f));
    atomicAdd(&acc[2], -logf(1.0f - nd + 1e-15f));
  }
}

__global__ void feat_loss_k(const float* __restrict__ x, const float* __restrict__ rm,
                            const float* __restrict__ db, const int* __restrict__ perm,
                            float* __restrict__ acc) {
  int m = blockIdx.x;
  int node = perm[m];
  float xx = 0.0f, rr = 0.0f, xr = 0.0f;
  for (int f = threadIdx.x; f < D_FEAT; f += blockDim.x) {
    float xv = x[(size_t)node * D_FEAT + f];
    float rv = rm[(size_t)m * D_FEAT + f] + db[f];
    xx += xv * xv; rr += rv * rv; xr += xv * rv;
  }
  __shared__ float s0[256], s1[256], s2[256];
  int t = threadIdx.x;
  s0[t] = xx; s1[t] = rr; s2[t] = xr;
  __syncthreads();
  for (int st = 128; st > 0; st >>= 1) {
    if (t < st) { s0[t] += s0[t + st]; s1[t] += s1[t + st]; s2[t] += s2[t + st]; }
    __syncthreads();
  }
  if (t == 0) {
    float nx = sqrtf(s0[0]); if (nx < 1e-12f) nx = 1e-12f;
    float nr = sqrtf(s1[0]); if (nr < 1e-12f) nr = 1e-12f;
    float cosv = s2[0] / (nx * nr);
    float d = 1.0f - cosv;
    atomicAdd(&acc[0], d * d);
  }
}

__global__ void zero_acc_k(float* acc) {
  if (threadIdx.x < 4) acc[threadIdx.x] = 0.0f;
}
__global__ void finalize_k(const float* __restrict__ acc, float* __restrict__ out) {
  out[0] = acc[0] / (float)NMASK;
  out[1] = acc[1] / (float)NMASK + acc[2] / (float)NMASK;
}

// ---------------------------------------------------------------------------
// Host orchestration
// ---------------------------------------------------------------------------
static inline long cdivl(long a, long b) { return (a + b - 1) / b; }

extern "C" void kernel_launch(void* const* d_in, const int* in_sizes, int n_in,
                              void* d_out, int out_size, void* d_ws, size_t ws_size,
                              hipStream_t stream) {
  (void)in_sizes; (void)n_in; (void)out_size; (void)ws_size;
  const float* x    = (const float*)d_in[0];
  const int*   ei   = (const int*)d_in[1];
  const int*   srcI = ei;
  const int*   dstI = ei + E_EDGES;
  const int*   perm = (const int*)d_in[2];
  const int*   shuf = (const int*)d_in[3];
  const float* sW1 = (const float*)d_in[4];  const float* sb1 = (const float*)d_in[5];
  const float* sa  = (const float*)d_in[6];  const float* sW2 = (const float*)d_in[7];
  const float* sb2 = (const float*)d_in[8];
  const float* tW1 = (const float*)d_in[9];  const float* tb1 = (const float*)d_in[10];
  const float* ta  = (const float*)d_in[11]; const float* tW2 = (const float*)d_in[12];
  const float* tb2 = (const float*)d_in[13];
  const float* pW1 = (const float*)d_in[14]; const float* pb1 = (const float*)d_in[15];
  const float* pa  = (const float*)d_in[16]; const float* pW2 = (const float*)d_in[17];
  const float* pb2 = (const float*)d_in[18];
  const float* tpW1 = (const float*)d_in[19]; const float* tpb1 = (const float*)d_in[20];
  const float* tpa  = (const float*)d_in[21]; const float* tpW2 = (const float*)d_in[22];
  const float* tpb2 = (const float*)d_in[23];
  const float* dgiW = (const float*)d_in[24];
  const float* posT = (const float*)d_in[25];
  const float* negT = (const float*)d_in[26];
  const float* e2dW = (const float*)d_in[27];
  const float* dW   = (const float*)d_in[28];
  const float* db   = (const float*)d_in[29];
  float* out = (float*)d_out;

  // Bump allocator over workspace.
  char* w = (char*)d_ws;
  size_t off = 0;
  auto alloc = [&](size_t bytes) -> void* {
    void* p = w + off;
    off = (off + bytes + 255) & ~(size_t)255;
    return p;
  };
  bf16_t* A_pos = (bf16_t*)alloc((size_t)N_NODES * DPAD * 2);
  bf16_t* A_neg = (bf16_t*)alloc((size_t)N_NODES * DPAD * 2);
  bf16_t* sW1t  = (bf16_t*)alloc((size_t)H_DIM * DPAD * 2);   // (H x DPAD) transposed
  bf16_t* tW1t  = (bf16_t*)alloc((size_t)H_DIM * DPAD * 2);
  bf16_t* sW2t  = (bf16_t*)alloc((size_t)L_DIM * H_DIM * 2);  // (L x H)
  bf16_t* tW2t  = (bf16_t*)alloc((size_t)L_DIM * H_DIM * 2);
  bf16_t* pW1t  = (bf16_t*)alloc((size_t)P_DIM * L_DIM * 2);  // (P x L)
  bf16_t* pW2t  = (bf16_t*)alloc((size_t)L_DIM * P_DIM * 2);  // (L x P)
  bf16_t* tpW1t = (bf16_t*)alloc((size_t)P_DIM * L_DIM * 2);
  bf16_t* tpW2t = (bf16_t*)alloc((size_t)L_DIM * P_DIM * 2);
  bf16_t* e2dt  = (bf16_t*)alloc((size_t)L_DIM * L_DIM * 2);  // (L x L)
  bf16_t* dWt   = (bf16_t*)alloc((size_t)D_FEAT * L_DIM * 2); // (D x L)
  float*  XW    = (float*)alloc((size_t)N_NODES * H_DIM * 4);
  float*  AGG1  = (float*)alloc((size_t)N_NODES * H_DIM * 4);
  bf16_t* H1b   = (bf16_t*)alloc((size_t)N_NODES * H_DIM * 2);
  float*  Z     = (float*)alloc((size_t)N_NODES * L_DIM * 4);
  float*  AGG2  = (float*)alloc((size_t)N_NODES * L_DIM * 4);
  float*  repPf = (float*)alloc((size_t)N_NODES * L_DIM * 4);
  bf16_t* repPb = (bf16_t*)alloc((size_t)N_NODES * L_DIM * 2);
  float*  repNf = (float*)alloc((size_t)N_NODES * L_DIM * 4);
  bf16_t* repNb = (bf16_t*)alloc((size_t)N_NODES * L_DIM * 2);
  float*  deg   = (float*)alloc((size_t)N_NODES * 4);
  float*  dinv  = (float*)alloc((size_t)N_NODES * 4);
  int*    n2m   = (int*)alloc((size_t)N_NODES * 4);
  bf16_t* M_in  = (bf16_t*)alloc((size_t)NMASK * L_DIM * 2);
  float*  P1    = (float*)alloc((size_t)NMASK * P_DIM * 4);
  bf16_t* P1b   = (bf16_t*)alloc((size_t)NMASK * P_DIM * 2);
  float*  posz  = (float*)alloc((size_t)NMASK * L_DIM * 4);
  float*  negz  = (float*)alloc((size_t)NMASK * L_DIM * 4);
  float*  svec  = (float*)alloc((size_t)L_DIM * 4);
  float*  wsv   = (float*)alloc((size_t)L_DIM * 4);
  float*  recL  = (float*)alloc((size_t)N_NODES * L_DIM * 4);
  bf16_t* recLb = (bf16_t*)alloc((size_t)N_NODES * L_DIM * 2);
  float*  recW  = (float*)alloc((size_t)N_NODES * D_FEAT * 4);
  float*  rmask = (float*)alloc((size_t)NMASK * D_FEAT * 4);
  float*  acc   = (float*)alloc(4 * 4);

  const int TB = 256;
  auto gsz = [&](long n) { long g = cdivl(n, TB); return (int)(g > 16384 ? 16384 : g); };

  hipLaunchKernelGGL(zero_acc_k, dim3(1), dim3(64), 0, stream, acc);

  // --- masking inputs (bf16, K padded to DPAD) ---
  hipLaunchKernelGGL(convert_pad_k, dim3(gsz((long)N_NODES * DPAD)), dim3(TB), 0, stream,
                     x, A_pos, (long)N_NODES, D_FEAT, DPAD);
  hipLaunchKernelGGL(convert_posmask_k, dim3(cdivl(D_FEAT, TB), NMASK), dim3(TB), 0, stream,
                     x, perm, posT, A_pos);
  hipLaunchKernelGGL(build_neg_k, dim3(cdivl(DPAD, TB), N_NODES), dim3(TB), 0, stream,
                     x, perm, shuf, negT, A_neg);

  // --- weight conversion to transposed bf16 ---
  hipLaunchKernelGGL(convert_w_t_k, dim3(gsz((long)H_DIM * DPAD)), dim3(TB), 0, stream, sW1, sW1t, D_FEAT, H_DIM, DPAD);
  hipLaunchKernelGGL(convert_w_t_k, dim3(gsz((long)H_DIM * DPAD)), dim3(TB), 0, stream, tW1, tW1t, D_FEAT, H_DIM, DPAD);
  hipLaunchKernelGGL(convert_w_t_k, dim3(gsz((long)L_DIM * H_DIM)), dim3(TB), 0, stream, sW2, sW2t, H_DIM, L_DIM, H_DIM);
  hipLaunchKernelGGL(convert_w_t_k, dim3(gsz((long)L_DIM * H_DIM)), dim3(TB), 0, stream, tW2, tW2t, H_DIM, L_DIM, H_DIM);
  hipLaunchKernelGGL(convert_w_t_k, dim3(gsz((long)P_DIM * L_DIM)), dim3(TB), 0, stream, pW1, pW1t, L_DIM, P_DIM, L_DIM);
  hipLaunchKernelGGL(convert_w_t_k, dim3(gsz((long)L_DIM * P_DIM)), dim3(TB), 0, stream, pW2, pW2t, P_DIM, L_DIM, P_DIM);
  hipLaunchKernelGGL(convert_w_t_k, dim3(gsz((long)P_DIM * L_DIM)), dim3(TB), 0, stream, tpW1, tpW1t, L_DIM, P_DIM, L_DIM);
  hipLaunchKernelGGL(convert_w_t_k, dim3(gsz((long)L_DIM * P_DIM)), dim3(TB), 0, stream, tpW2, tpW2t, P_DIM, L_DIM, P_DIM);
  hipLaunchKernelGGL(convert_w_t_k, dim3(gsz((long)L_DIM * L_DIM)), dim3(TB), 0, stream, e2dW, e2dt, L_DIM, L_DIM, L_DIM);
  hipLaunchKernelGGL(convert_w_t_k, dim3(gsz((long)D_FEAT * L_DIM)), dim3(TB), 0, stream, dW, dWt, L_DIM, D_FEAT, L_DIM);

  // --- degrees / normalization / mask map ---
  hipLaunchKernelGGL(init_deg_k, dim3(cdivl(N_NODES, TB)), dim3(TB), 0, stream, deg);
  hipLaunchKernelGGL(deg_edges_k, dim3(cdivl(E_EDGES, TB)), dim3(TB), 0, stream, dstI, deg);
  hipLaunchKernelGGL(dinv_k, dim3(cdivl(N_NODES, TB)), dim3(TB), 0, stream, deg, dinv);
  hipLaunchKernelGGL(n2m_init_k, dim3(cdivl(N_NODES, TB)), dim3(TB), 0, stream, n2m);
  hipLaunchKernelGGL(n2m_set_k, dim3(cdivl(NMASK, TB)), dim3(TB), 0, stream, perm, n2m);

  // --- encoder (pos then neg, reusing buffers) ---
  const bf16_t* Ain[2]  = { A_pos, A_neg };
  const bf16_t* W1t[2]  = { sW1t, tW1t };
  const bf16_t* W2t[2]  = { sW2t, tW2t };
  const float*  b1[2]   = { sb1, tb1 };
  const float*  b2[2]   = { sb2, tb2 };
  const float*  aa[2]   = { sa, ta };
  float*  repF[2] = { repPf, repNf };
  bf16_t* repB[2] = { repPb, repNb };
  for (int br = 0; br < 2; ++br) {
    hipLaunchKernelGGL(gemm_bf16_k, dim3(cdivl(H_DIM, 128), cdivl(N_NODES, 64)), dim3(256), 0,
                       stream, Ain[br], W1t[br], XW, N_NODES, H_DIM, DPAD);
    hipLaunchKernelGGL(agg_init_k, dim3(cdivl(H_DIM, TB), N_NODES), dim3(TB), 0, stream, XW, dinv, AGG1, H_DIM);
    hipLaunchKernelGGL(agg_edges_k, dim3(cdivl(H_DIM, TB), E_EDGES), dim3(TB), 0, stream, XW, srcI, dstI, dinv, AGG1, H_DIM);
    hipLaunchKernelGGL(bias_prelu_k, dim3(gsz((long)N_NODES * H_DIM)), dim3(TB), 0, stream,
                       AGG1, b1[br], aa[br], (float*)nullptr, H1b, H_DIM, (long)N_NODES * H_DIM);
    hipLaunchKernelGGL(gemm_bf16_k, dim3(cdivl(L_DIM, 128), cdivl(N_NODES, 64)), dim3(256), 0,
                       stream, H1b, W2t[br], Z, N_NODES, L_DIM, H_DIM);
    hipLaunchKernelGGL(agg_init_k, dim3(cdivl(L_DIM, TB), N_NODES), dim3(TB), 0, stream, Z, dinv, AGG2, L_DIM);
    hipLaunchKernelGGL(agg_edges_k, dim3(cdivl(L_DIM, TB), E_EDGES), dim3(TB), 0, stream, Z, srcI, dstI, dinv, AGG2, L_DIM);
    hipLaunchKernelGGL(bias_prelu_k, dim3(gsz((long)N_NODES * L_DIM)), dim3(TB), 0, stream,
                       AGG2, b2[br], aa[br], repF[br], repB[br], L_DIM, (long)N_NODES * L_DIM);
  }

  // --- projection heads on masked rows ---
  const bf16_t* pjW1[2] = { pW1t, tpW1t };
  const bf16_t* pjW2[2] = { pW2t, tpW2t };
  const float*  pjb1[2] = { pb1, tpb1 };
  const float*  pjb2[2] = { pb2, tpb2 };
  const float*  pja[2]  = { pa, tpa };
  float* zout[2] = { posz, negz };
  for (int br = 0; br < 2; ++br) {
    hipLaunchKernelGGL(gather_rows_k, dim3(cdivl(L_DIM, TB), NMASK), dim3(TB), 0, stream,
                       repB[br], perm, M_in, L_DIM);
    hipLaunchKernelGGL(gemm_bf16_k, dim3(cdivl(P_DIM, 128), cdivl(NMASK, 64)), dim3(256), 0,
                       stream, M_in, pjW1[br], P1, NMASK, P_DIM, L_DIM);
    hipLaunchKernelGGL(bias_prelu_k, dim3(gsz((long)NMASK * P_DIM)), dim3(TB), 0, stream,
                       P1, pjb1[br], pja[br], (float*)nullptr, P1b, P_DIM, (long)NMASK * P_DIM);
    hipLaunchKernelGGL(gemm_bf16_k, dim3(cdivl(L_DIM, 128), cdivl(NMASK, 64)), dim3(256), 0,
                       stream, P1b, pjW2[br], zout[br], NMASK, L_DIM, P_DIM);
    hipLaunchKernelGGL(add_bias_k, dim3(gsz((long)NMASK * L_DIM)), dim3(TB), 0, stream,
                       zout[br], pjb2[br], L_DIM, (long)NMASK * L_DIM);
  }

  // --- DGI loss ---
  hipLaunchKernelGGL(colmean_sigmoid_k, dim3(L_DIM), dim3(256), 0, stream, posz, svec, NMASK, L_DIM);
  hipLaunchKernelGGL(matvec128_k, dim3(1), dim3(L_DIM), 0, stream, dgiW, svec, wsv);
  hipLaunchKernelGGL(dgi_rows_k, dim3(NMASK), dim3(L_DIM), 0, stream, posz, negz, wsv, acc);

  // --- decoder / reconstruction (mask rows only) ---
  hipLaunchKernelGGL(gemm_bf16_k, dim3(cdivl(L_DIM, 128), cdivl(N_NODES, 64)), dim3(256), 0,
                     stream, repPb, e2dt, recL, N_NODES, L_DIM, L_DIM);
  hipLaunchKernelGGL(zero_mask_rows_k, dim3(NMASK), dim3(L_DIM), 0, stream, recL, perm);
  hipLaunchKernelGGL(convert_f2b_k, dim3(gsz((long)N_NODES * L_DIM)), dim3(TB), 0, stream,
                     recL, recLb, (long)N_NODES * L_DIM);
  hipLaunchKernelGGL(gemm_bf16_k, dim3(cdivl(D_FEAT, 128), cdivl(N_NODES, 64)), dim3(256), 0,
                     stream, recLb, dWt, recW, N_NODES, D_FEAT, L_DIM);
  hipLaunchKernelGGL(rec_mask_init_k, dim3(cdivl(D_FEAT, TB), NMASK), dim3(TB), 0, stream,
                     recW, dinv, perm, rmask);
  hipLaunchKernelGGL(rec_mask_edges_k, dim3(cdivl(D_FEAT, TB), E_EDGES), dim3(TB), 0, stream,
                     recW, srcI, dstI, n2m, dinv, rmask);

  // --- feature (cosine) loss + finalize ---
  hipLaunchKernelGGL(feat_loss_k, dim3(NMASK), dim3(256), 0, stream, x, rmask, db, perm, acc);
  hipLaunchKernelGGL(finalize_k, dim3(1), dim3(1), 0, stream, acc, out);
}
